// GearNetIEConv_46428596470372
// MI455X (gfx1250) — matrix operand order repaired and checked by reading, not compiled, not run on hardware
//
#include <hip/hip_runtime.h>

// ---------------------------------------------------------------------------
// GearNet-IEConv forward for MI455X (gfx1250, wave32, WMMA bf16 w/ f32 accum)
// - B-operands pre-swizzled in LDS to fragment-major v16bf (ds_load_b128 x2)
// - edge-tile staging via GLOBAL_LOAD_ASYNC_TO_LDS_B128 + s_wait_asynccnt
// ---------------------------------------------------------------------------
typedef __bf16 bf16;
typedef __attribute__((ext_vector_type(16))) __bf16 v16bf;
typedef __attribute__((ext_vector_type(8)))  float  v8f;

#define BN_EPS 1e-5f
#define LN_EPS 1e-5f

#define WMMA_BF16(A, B, C) \
  __builtin_amdgcn_wmma_f32_16x16x32_bf16(false, (A), false, (B), (short)0, (C), false, false)

// ---- CDNA5 async global->LDS copy (16B per lane), ASYNCcnt-tracked --------
__device__ __forceinline__ void async_copy_b128(unsigned lds_off, const void* gptr) {
  unsigned long long ga = (unsigned long long)(uintptr_t)gptr;
  asm volatile("global_load_async_to_lds_b128 %0, %1, off"
               :: "v"(lds_off), "v"(ga) : "memory");
}
__device__ __forceinline__ void wait_async0() {
  asm volatile("s_wait_asynccnt 0" ::: "memory");
}
__device__ __forceinline__ unsigned lds_off_of(const void* p) {
  // generic shared-memory address: addr[31:0] is the LDS byte offset
  return (unsigned)(uintptr_t)p;
}

// ---- CDNA5 wave32 WMMA A-fragment loader (ISA 7.12.2) ---------------------
// A (16x32 bf16): lanes 0-15 hold K {0..7, 16..23}; lanes 16-31 hold K {8..15, 24..31}
// Per lane this is two contiguous 16B runs -> compiler emits 2x ds_load_b128.
__device__ __forceinline__ v16bf load_a_frag(const bf16* a, int stride) {
  int lane = threadIdx.x & 31;
  int m = lane & 15, hi = lane >> 4;
  v16bf f;
#pragma unroll
  for (int v = 0; v < 8; ++v) {
    int kb = ((v < 4) ? 0 : 16) + hi * 8 + (v & 3) * 2;
    f[2 * v]     = a[m * stride + kb];
    f[2 * v + 1] = a[m * stride + kb + 1];
  }
  return f;
}

// B-fragment slot map (32x16 bf16): lane = hi*16 + n; slot s -> (v=s>>1, e=s&1)
// element = B[k = (s<8 ? 0 : 16) + hi*8 + (s&7)][n].  B tiles are PRE-PACKED into v16bf[32].
__device__ __forceinline__ void pack_b_slot(int idx, int& lane, int& k, int& n) {
  lane = (idx >> 4) & 31;
  int s = idx & 15;
  int hi = lane >> 4;
  n = lane & 15;
  k = ((s < 8) ? 0 : 16) + hi * 8 + (s & 7);
}
// C/D (16x16 f32): lane holds col = lane&15, rows m = (lane>>4)*8 + r, r=0..7

// ---------------------------------------------------------------------------
__global__ void k_zero_f(float* p, size_t n) {
  size_t i = (size_t)blockIdx.x * blockDim.x + threadIdx.x;
  if (i < n) p[i] = 0.f;
}

__global__ void k_copy_f(const float* s, float* d, size_t n) {
  size_t i = (size_t)blockIdx.x * blockDim.x + threadIdx.x;
  if (i < n) d[i] = s[i];
}

// ---- local reference frames per node --------------------------------------
__device__ __forceinline__ void norm3(float& x, float& y, float& z) {
  float n = sqrtf(x * x + y * y + z * z);
  float inv = 1.f / fmaxf(n, 1e-12f);
  x *= inv; y *= inv; z *= inv;
}

__global__ void k_frames(const float* __restrict__ pos, float* __restrict__ frames, int N) {
  int i = blockIdx.x * blockDim.x + threadIdx.x;
  if (i >= N) return;
  float u0x, u0y, u0z;
  if (i == 0) { u0x = u0y = u0z = 1.f; }
  else { u0x = pos[i*3+0]-pos[(i-1)*3+0]; u0y = pos[i*3+1]-pos[(i-1)*3+1]; u0z = pos[i*3+2]-pos[(i-1)*3+2]; }
  norm3(u0x, u0y, u0z);
  float bx, by, bz, nx, ny, nz;
  if (i < N - 1) {
    float u1x = pos[(i+1)*3+0]-pos[i*3+0], u1y = pos[(i+1)*3+1]-pos[i*3+1], u1z = pos[(i+1)*3+2]-pos[i*3+2];
    norm3(u1x, u1y, u1z);
    bx = u0x - u1x; by = u0y - u1y; bz = u0z - u1z;
    nx = u0y*u1z - u0z*u1y; ny = u0z*u1x - u0x*u1z; nz = u0x*u1y - u0y*u1x;
  } else {
    bx = by = bz = 1.f; nx = ny = nz = 1.f;
  }
  norm3(bx, by, bz); norm3(nx, ny, nz);
  float cx = by*nz - bz*ny, cy = bz*nx - bx*nz, cz = bx*ny - by*nx;
  frames[i*9+0] = bx; frames[i*9+1] = nx; frames[i*9+2] = cx;
  frames[i*9+3] = by; frames[i*9+4] = ny; frames[i*9+5] = cy;
  frames[i*9+6] = bz; frames[i*9+7] = nz; frames[i*9+8] = cz;
}

// ---- edge features (14, padded to 16, bf16) + degree ----------------------
__global__ void k_edgefeat(const float* __restrict__ pos, const float* __restrict__ frames,
                           const int* __restrict__ ei, bf16* __restrict__ ef,
                           float* __restrict__ deg, int E) {
  int e = blockIdx.x * blockDim.x + threadIdx.x;
  if (e >= E) return;
  int s = ei[e], d = ei[E + e];
  float dx = pos[d*3+0]-pos[s*3+0], dy = pos[d*3+1]-pos[s*3+1], dz = pos[d*3+2]-pos[s*3+2];
  float t[3], r[3];
#pragma unroll
  for (int k = 0; k < 3; ++k) {
    float f0 = frames[s*9+0*3+k], f1 = frames[s*9+1*3+k], f2 = frames[s*9+2*3+k];
    t[k] = f0*dx + f1*dy + f2*dz;
    r[k] = f0*frames[d*9+0*3+k] + f1*frames[d*9+1*3+k] + f2*frames[d*9+2*3+k];
  }
  float delta = fabsf((float)(s - d)) * (1.f / 6.f);
  float feat[16];
#pragma unroll
  for (int k = 0; k < 3; ++k) { feat[k] = t[k]; feat[3+k] = r[k]; feat[7+k] = 1.f - 2.f*fabsf(t[k]); feat[10+k] = 1.f - 2.f*fabsf(r[k]); }
  feat[6] = delta; feat[13] = 1.f - 2.f*fabsf(delta);
  feat[14] = 0.f; feat[15] = 0.f;
#pragma unroll
  for (int k = 0; k < 16; ++k) ef[(size_t)e*16 + k] = (bf16)feat[k];
  atomicAdd(&deg[d], 1.f);
}

// ---- per-layer prep: li = relu(bn_in(h)) as bf16; zero ie accumulator -----
__global__ void k_prep(const float* __restrict__ h, const float* __restrict__ bn_in,
                       bf16* __restrict__ li_bf, float* __restrict__ ie_accum, int N) {
  size_t gid = (size_t)blockIdx.x * blockDim.x + threadIdx.x;
  if (gid >= (size_t)N * 64) return;
  int c = (int)(gid & 63);
  float x = h[gid];
  float li = (x - bn_in[2*64+c]) * rsqrtf(bn_in[3*64+c] + BN_EPS) * bn_in[0*64+c] + bn_in[1*64+c];
  li_bf[gid] = (bf16)fmaxf(li, 0.f);
  if (c < 16) ie_accum[(gid >> 6) * 16 + c] = 0.f;
}

// ---- relation-bucketed scatter: updr[dst] += h[src] for edges of rel r ----
__global__ void k_scatter_rel(const float* __restrict__ h, const int* __restrict__ ei,
                              const int* __restrict__ erel, float* __restrict__ updr,
                              int r, int E) {
  size_t gid = (size_t)blockIdx.x * blockDim.x + threadIdx.x;
  if (gid >= (size_t)E * 64) return;
  int e = (int)(gid >> 6), c = (int)(gid & 63);
  if (erel[e] != r) return;
  atomicAdd(&updr[(size_t)ei[E + e] * 64 + c], h[(size_t)ei[e] * 64 + c]);
}

// ---- [N,64] @ [64,64] GEMM via WMMA bf16; 4 waves share packed B ----------
__global__ void __launch_bounds__(128)
k_gemm64(const float* __restrict__ A, const float* __restrict__ B,
         const float* __restrict__ bias0, const float* __restrict__ bias1,
         float* __restrict__ C, int nrows, int accumulate) {
  __shared__ bf16  sA[4][16][64];
  __shared__ v16bf sBf[8][32];          // frag f = kt*4+nt, per-lane packed
  int tid = threadIdx.x;
  int wave = tid >> 5, lane = tid & 31;
  int row0 = blockIdx.x * 64 + wave * 16;

  for (int idx = tid; idx < 4096; idx += 128) {
    int f = idx >> 9;
    int l2, k, n2; pack_b_slot(idx, l2, k, n2);
    int kt = f >> 2, nt = f & 3;
    ((bf16*)sBf)[idx] = (bf16)B[(kt * 32 + k) * 64 + nt * 16 + n2];
  }
  for (int idx = lane; idx < 16 * 64; idx += 32) {
    int rr = idx >> 6, cc = idx & 63;
    int gr = row0 + rr;
    sA[wave][rr][cc] = (gr < nrows) ? (bf16)A[(size_t)gr * 64 + cc] : (bf16)0.f;
  }
  __syncthreads();

  int n = lane & 15, hi = lane >> 4;
#pragma unroll
  for (int nt = 0; nt < 4; ++nt) {
    v8f acc = {};
#pragma unroll
    for (int kt = 0; kt < 2; ++kt) {
      v16bf a = load_a_frag(&sA[wave][0][kt * 32], 64);
      v16bf b = sBf[kt * 4 + nt][lane];
      acc = WMMA_BF16(a, b, acc);
    }
    int col = nt * 16 + n;
    float bia = (bias0 ? bias0[col] : 0.f) + (bias1 ? bias1[col] : 0.f);
#pragma unroll
    for (int r = 0; r < 8; ++r) {
      int grow = row0 + hi * 8 + r;
      if (grow < nrows) {
        float v = acc[r] + bia;
        size_t o = (size_t)grow * 64 + col;
        if (accumulate) C[o] += v; else C[o] = v;
      }
    }
  }
}

// ---- the heavy edge pipeline: 21 WMMAs per 16-edge tile -------------------
__global__ void __launch_bounds__(128)
k_edge_ie(const bf16* __restrict__ li_bf, const bf16* __restrict__ ef,
          const int* __restrict__ ei,
          const float* __restrict__ W_l1, const float* __restrict__ b_l1,
          const float* __restrict__ W_k1, const float* __restrict__ b_k1,
          const float* __restrict__ W_k2, const float* __restrict__ b_k2,
          const float* __restrict__ bn_msg, float* __restrict__ ie_accum,
          int E, int nTiles) {
  __shared__ v16bf sWl1f[2][32];        // kt tiles of W_l1 (K=64 -> 2)
  __shared__ v16bf sWk1f[2][32];        // nt tiles of W_k1 (rows 14..31 zero)
  __shared__ v16bf sWk2f[17][32];       // jt tiles of W_k2
  __shared__ float sbl1[16], sbk1[32], sbk2[272], sbnm[4][16];
  __shared__ bf16 sAli[4][16][64];
  __shared__ bf16 sAef[4][16][32];
  __shared__ bf16 sKf[4][16][32];
  __shared__ __attribute__((aligned(16))) float sMt[4][16][16];  // [j][e] transposed
  __shared__ int  sSrc[4][16], sDst[4][16], sVal[4][16];

  int tid = threadIdx.x;
  // stage weights fragment-major (power-of-two indexing only)
  for (int idx = tid; idx < 2 * 512; idx += 128) {
    int kt = idx >> 9;
    int l2, k, n2; pack_b_slot(idx, l2, k, n2);
    ((bf16*)sWl1f)[idx] = (bf16)W_l1[(kt * 32 + k) * 16 + n2];
  }
  for (int idx = tid; idx < 2 * 512; idx += 128) {
    int nt = idx >> 9;
    int l2, k, n2; pack_b_slot(idx, l2, k, n2);
    ((bf16*)sWk1f)[idx] = (k < 14) ? (bf16)W_k1[k * 32 + nt * 16 + n2] : (bf16)0.f;
  }
  for (int idx = tid; idx < 17 * 512; idx += 128) {
    int jt = idx >> 9;
    int l2, k, n2; pack_b_slot(idx, l2, k, n2);
    ((bf16*)sWk2f)[idx] = (bf16)W_k2[k * 272 + jt * 16 + n2];
  }
  if (tid < 16)  sbl1[tid] = b_l1[tid];
  if (tid < 32)  sbk1[tid] = b_k1[tid];
  for (int idx = tid; idx < 272; idx += 128) sbk2[idx] = b_k2[idx];
  if (tid < 64)  sbnm[tid >> 4][tid & 15] = bn_msg[tid];
  // zero the always-zero K=16..31 half of the ef A-tile ONCE (async writes
  // only touch K=0..15)
  for (int idx = tid; idx < 4 * 16 * 16; idx += 128) {
    int w = idx >> 8, rr = (idx >> 4) & 15, cc = idx & 15;
    sAef[w][rr][16 + cc] = (bf16)0.f;
  }
  __syncthreads();

  int wave = tid >> 5, lane = tid & 31;
  int n = lane & 15, hi = lane >> 4;
  int tilesPerPass = gridDim.x * 4;
  int nIter = (nTiles + tilesPerPass - 1) / tilesPerPass;
  unsigned sAliBase = lds_off_of(&sAli[wave][0][0]);
  unsigned sAefBase = lds_off_of(&sAef[wave][0][0]);

  for (int it = 0; it < nIter; ++it) {
    int tile = it * tilesPerPass + blockIdx.x * 4 + wave;
    int e0 = tile * 16;

    // prefetch next iteration's edge window (global_prefetch_b8)
    {
      int tnext = (it + 1) * tilesPerPass + blockIdx.x * 4 + wave;
      int en = tnext * 16;
      if (tnext < nTiles && en < E) {
        __builtin_prefetch(&ef[(size_t)en * 16], 0, 1);
        __builtin_prefetch(&ei[en], 0, 1);
        __builtin_prefetch(&ei[E + en], 0, 1);
      }
    }

    bool allvalid = (tile < nTiles) && (e0 + 16 <= E);
    if (lane < 16) {
      int e = e0 + lane;
      bool v = (e < E) && (tile < nTiles);
      sSrc[wave][lane] = v ? ei[e] : 0;
      sDst[wave][lane] = v ? ei[E + e] : 0;
      sVal[wave][lane] = v ? 1 : 0;
    }
    __syncthreads();

    if (allvalid) {
      // async DMA staging: 4x b128 for the gathered li tile (2KB),
      // 1x b128 for the edge-feature tile (512B); per-lane addresses.
      int row = lane >> 1, half = lane & 1;
      int src = sSrc[wave][row];
      const char* gsrc = (const char*)li_bf + ((size_t)src * 128 + half * 64);
      unsigned lrow = sAliBase + (unsigned)(row * 128 + half * 64);
#pragma unroll
      for (int q = 0; q < 4; ++q)
        async_copy_b128(lrow + q * 16, gsrc + q * 16);
      async_copy_b128(sAefBase + (unsigned)(row * 64 + half * 16),
                      (const char*)ef + ((size_t)(e0 + row) * 32 + half * 16));
      wait_async0();
    } else {
      int row = lane >> 1, half = lane & 1;
      int src = sSrc[wave][row];
      bool v = sVal[wave][row] != 0;
      int e = e0 + row;
#pragma unroll
      for (int c = 0; c < 32; ++c)
        sAli[wave][row][half * 32 + c] = v ? li_bf[(size_t)src * 64 + half * 32 + c] : (bf16)0.f;
#pragma unroll
      for (int c = 0; c < 8; ++c)
        sAef[wave][row][half * 8 + c] = (v && (half * 8 + c) < 16) ? ef[(size_t)e * 16 + half * 8 + c] : (bf16)0.f;
    }
    __syncthreads();

    // m = relu(bn_msg(li[src] @ W_l1 + b_l1)) -> LDS f32, stored [j][e]
    {
      v8f acc = {};
#pragma unroll
      for (int kt = 0; kt < 2; ++kt) {
        v16bf a = load_a_frag(&sAli[wave][0][kt * 32], 64);
        v16bf b = sWl1f[kt][lane];
        acc = WMMA_BF16(a, b, acc);
      }
#pragma unroll
      for (int r = 0; r < 8; ++r) {
        int mr = hi * 8 + r;                 // edge row
        float v = acc[r] + sbl1[n];          // n = hidden channel j
        v = (v - sbnm[2][n]) * rsqrtf(sbnm[3][n] + BN_EPS) * sbnm[0][n] + sbnm[1][n];
        sMt[wave][n][mr] = fmaxf(v, 0.f);    // transposed store
      }
    }
    // kf = relu(ef @ W_k1 + b_k1) -> LDS bf16 [16][32]
    {
      v16bf a = load_a_frag(&sAef[wave][0][0], 32);
#pragma unroll
      for (int nt = 0; nt < 2; ++nt) {
        v8f acc = {};
        v16bf b = sWk1f[nt][lane];
        acc = WMMA_BF16(a, b, acc);
        int col = nt * 16 + n;
#pragma unroll
        for (int r = 0; r < 8; ++r)
          sKf[wave][hi * 8 + r][col] = (bf16)fmaxf(acc[r] + sbk1[col], 0.f);
      }
    }
    __syncthreads();

    // ker = kf @ W_k2 + b_k2 (17 tiles; tile 0 = bias row of per-edge 16x16
    // kernel; tiles 1..16 contracted against m on VALU with float4 LDS reads)
    {
      v16bf a = load_a_frag(&sKf[wave][0][0], 32);   // reused across 17 WMMAs
      v8f out = {};
#pragma unroll
      for (int jt = 0; jt < 17; ++jt) {
        v16bf b = sWk2f[jt][lane];
        v8f acc = {};
        acc = WMMA_BF16(a, b, acc);
        if (jt == 0) {
#pragma unroll
          for (int r = 0; r < 8; ++r) out[r] = acc[r] + sbk2[n];
        } else {
          const float4* mp = (const float4*)&sMt[wave][jt - 1][hi * 8];
          float mv[8];
          *(float4*)&mv[0] = mp[0];
          *(float4*)&mv[4] = mp[1];
          float bia = sbk2[jt * 16 + n];
#pragma unroll
          for (int r = 0; r < 8; ++r) out[r] += (acc[r] + bia) * mv[r];
        }
      }
      // scatter-mean numerator
#pragma unroll
      for (int r = 0; r < 8; ++r) {
        int er = hi * 8 + r;
        if (sVal[wave][er])
          atomicAdd(&ie_accum[(size_t)sDst[wave][er] * 16 + n], out[r]);
      }
    }
    __syncthreads();
  }
}

// ---- node-wise epilogue: relu(rel) + bn_out(bn_upd-relu-mean @ W_l2) + h, LN
__global__ void k_finalize(const float* __restrict__ rel_accum, const float* __restrict__ ie_accum,
                           const float* __restrict__ deg,
                           const float* __restrict__ W_l2, const float* __restrict__ b_l2,
                           const float* __restrict__ bn_upd, const float* __restrict__ bn_out,
                           const float* __restrict__ ln_g, const float* __restrict__ ln_b,
                           float* __restrict__ h, float* __restrict__ out_final, int N) {
  int nid = blockIdx.x;
  int c = threadIdx.x;  // 64 threads
  __shared__ float sU[16];
  __shared__ float sS[64], sS2[64];
  if (nid >= N) return;
  float dg = fmaxf(deg[nid], 1.f);
  if (c < 16) {
    float v = ie_accum[(size_t)nid * 16 + c] / dg;
    v = (v - bn_upd[2*16+c]) * rsqrtf(bn_upd[3*16+c] + BN_EPS) * bn_upd[0*16+c] + bn_upd[1*16+c];
    sU[c] = fmaxf(v, 0.f);
  }
  __syncthreads();
  float ie = b_l2[c];
#pragma unroll
  for (int j = 0; j < 16; ++j) ie += sU[j] * W_l2[j * 64 + c];
  ie = (ie - bn_out[2*64+c]) * rsqrtf(bn_out[3*64+c] + BN_EPS) * bn_out[0*64+c] + bn_out[1*64+c];
  float rel = fmaxf(rel_accum[(size_t)nid * 64 + c], 0.f);
  float hid = rel + ie + h[(size_t)nid * 64 + c];
  sS[c] = hid; sS2[c] = hid * hid;
  __syncthreads();
  for (int s = 32; s > 0; s >>= 1) {
    if (c < s) { sS[c] += sS[c + s]; sS2[c] += sS2[c + s]; }
    __syncthreads();
  }
  float mu = sS[0] * (1.f / 64.f);
  float var = sS2[0] * (1.f / 64.f) - mu * mu;
  float hn = (hid - mu) * rsqrtf(var + LN_EPS) * ln_g[c] + ln_b[c];
  h[(size_t)nid * 64 + c] = hn;
  if (out_final) out_final[(size_t)nid * 64 + c] = hn;
}

// ---------------------------------------------------------------------------
extern "C" void kernel_launch(void* const* d_in, const int* in_sizes, int n_in,
                              void* d_out, int out_size, void* d_ws, size_t ws_size,
                              hipStream_t stream) {
  const float* x      = (const float*)d_in[0];
  const float* pos    = (const float*)d_in[1];
  const int*   ei     = (const int*)  d_in[2];
  const int*   erel   = (const int*)  d_in[3];
  const float* W_rel  = (const float*)d_in[4];
  const float* b_rel  = (const float*)d_in[5];
  const float* W_self = (const float*)d_in[6];
  const float* b_self = (const float*)d_in[7];
  const float* W_l1   = (const float*)d_in[8];
  const float* b_l1   = (const float*)d_in[9];
  const float* W_k1   = (const float*)d_in[10];
  const float* b_k1   = (const float*)d_in[11];
  const float* W_k2   = (const float*)d_in[12];
  const float* b_k2   = (const float*)d_in[13];
  const float* W_l2   = (const float*)d_in[14];
  const float* b_l2   = (const float*)d_in[15];
  const float* bn_in  = (const float*)d_in[16];
  const float* bn_msg = (const float*)d_in[17];
  const float* bn_upd = (const float*)d_in[18];
  const float* bn_out = (const float*)d_in[19];
  const float* ln_g   = (const float*)d_in[20];
  const float* ln_b   = (const float*)d_in[21];

  const int N = in_sizes[0] / 64;
  const int E = in_sizes[3];
  const int R = 7, L = 3;

  // ---- workspace carve (~125 MB) ----
  size_t off = 0;
  auto carve = [&](size_t bytes) -> void* {
    void* p = (char*)d_ws + off;
    off += (bytes + 255) & ~(size_t)255;
    return p;
  };
  float* frames    = (float*)carve((size_t)N * 9 * 4);
  float* deg       = (float*)carve((size_t)N * 4);
  bf16*  ef        = (bf16*) carve((size_t)E * 16 * 2);
  bf16*  li_bf     = (bf16*) carve((size_t)N * 64 * 2);
  float* h         = (float*)carve((size_t)N * 64 * 4);
  float* rel_accum = (float*)carve((size_t)N * 64 * 4);
  float* updr      = (float*)carve((size_t)N * 64 * 4);
  float* ie_accum  = (float*)carve((size_t)N * 16 * 4);
  (void)ws_size; (void)n_in; (void)out_size;

  auto blocks = [](size_t n, int bs) { return (unsigned)((n + bs - 1) / bs); };

  k_zero_f<<<blocks(N, 256), 256, 0, stream>>>(deg, (size_t)N);
  k_copy_f<<<blocks((size_t)N * 64, 256), 256, 0, stream>>>(x, h, (size_t)N * 64);
  k_frames<<<blocks(N, 256), 256, 0, stream>>>(pos, frames, N);
  k_edgefeat<<<blocks(E, 256), 256, 0, stream>>>(pos, frames, ei, ef, deg, E);

  const int gemmGrid = (N + 63) / 64;
  const int nTiles = (E + 15) / 16;

  for (int l = 0; l < L; ++l) {
    k_prep<<<blocks((size_t)N * 64, 256), 256, 0, stream>>>(
        h, bn_in + (size_t)l * 4 * 64, li_bf, ie_accum, N);

    // rel_accum = h @ W_self + b_self + b_rel
    k_gemm64<<<gemmGrid, 128, 0, stream>>>(
        h, W_self + (size_t)l * 64 * 64, b_self + (size_t)l * 64,
        b_rel + (size_t)l * 64, rel_accum, N, 0);

    // + sum_r (scatter_r(h) @ W_rel[r])
    for (int r = 0; r < R; ++r) {
      k_zero_f<<<blocks((size_t)N * 64, 256), 256, 0, stream>>>(updr, (size_t)N * 64);
      k_scatter_rel<<<blocks((size_t)E * 64, 256), 256, 0, stream>>>(h, ei, erel, updr, r, E);
      k_gemm64<<<gemmGrid, 128, 0, stream>>>(
          updr, W_rel + ((size_t)l * 448 + (size_t)r * 64) * 64,
          nullptr, nullptr, rel_accum, N, 1);
    }

    // IEConv edge pipeline (WMMA + async-to-LDS staging)
    k_edge_ie<<<1024, 128, 0, stream>>>(
        li_bf, ef, ei,
        W_l1 + (size_t)l * 64 * 16, b_l1 + (size_t)l * 16,
        W_k1 + (size_t)l * 14 * 32, b_k1 + (size_t)l * 32,
        W_k2 + (size_t)l * 32 * 272, b_k2 + (size_t)l * 272,
        bn_msg + (size_t)l * 4 * 16, ie_accum, E, nTiles);

    k_finalize<<<N, 64, 0, stream>>>(
        rel_accum, ie_accum, deg,
        W_l2 + (size_t)l * 16 * 64, b_l2 + (size_t)l * 64,
        bn_upd + (size_t)l * 4 * 16, bn_out + (size_t)l * 4 * 64,
        ln_g + (size_t)l * 64, ln_b + (size_t)l * 64,
        h, (l == L - 1) ? (float*)d_out : nullptr, N);
  }
}
